// MobileFormerBlock_64819646431348
// MI455X (gfx1250) — matrix operand order repaired and checked by reading, not compiled, not run on hardware
//
#include <hip/hip_runtime.h>

// ---------------------------------------------------------------------------
// MobileFormer block for gfx1250 (MI455X).
// Heavy GEMMs (pointwise convs) run on v_wmma_f32_16x16x32_bf16 with f32
// accumulation; B-tile staging uses async global->LDS copies (ASYNCcnt) when
// the toolchain exposes the builtin. Everything else (token path, depthwise
// conv, attentions) is bandwidth-bound VALU code.
// ---------------------------------------------------------------------------

typedef __bf16 bf16;
typedef __attribute__((ext_vector_type(16))) __bf16 v16bf;
typedef __attribute__((ext_vector_type(8)))  float  v8f;

#define TN 6
#define BN 64
#define CT 192
#define CIN 128
#define HW 1024
#define OUT_ELEMS (BN * CIN * HW)   // 8,388,608

// --- optional CDNA5 async global->LDS copy path ----------------------------
#if defined(__has_builtin)
#  if __has_builtin(__builtin_amdgcn_global_load_async_to_lds_b128) && \
      __has_builtin(__builtin_amdgcn_s_wait_asynccnt)
#    define USE_ASYNC_LDS 1
#  endif
#endif
#ifndef USE_ASYNC_LDS
#  define USE_ASYNC_LDS 0
#endif

#if USE_ASYNC_LDS
// Builtin signature (from hipcc diagnostic): arg0 = int4 addrspace(1)* (global
// source, non-const), arg1 = int4 addrspace(3)* (LDS dest), arg2/3 = imm ints.
typedef int v4i_t __attribute__((ext_vector_type(4)));
typedef __attribute__((address_space(1))) v4i_t* gv4i_ptr;
typedef __attribute__((address_space(3))) v4i_t* lv4i_ptr;
#endif

__device__ __forceinline__ float hsig(float x) {
    return fminf(fmaxf((x + 3.0f) * (1.0f / 6.0f), 0.0f), 1.0f);
}

// ---------------------------------------------------------------------------
// K0: f32 -> bf16 convert (feat, pw1_w, pw2_w)
// ---------------------------------------------------------------------------
__global__ void cvt_f32_bf16(const float* __restrict__ src, bf16* __restrict__ dst, int n) {
    for (int i = blockIdx.x * blockDim.x + threadIdx.x; i < n; i += gridDim.x * blockDim.x)
        dst[i] = (bf16)src[i];
}

// ---------------------------------------------------------------------------
// Block-wide layernorm over 192 elements (256-thread block).
// ---------------------------------------------------------------------------
__device__ __forceinline__ void block_ln_192(const float* r, float* red, int tid,
                                             const float* g, const float* bb,
                                             float* outp) {
    float v = (tid < CT) ? r[tid] : 0.0f;
    red[tid] = v; __syncthreads();
    for (int s = 128; s > 0; s >>= 1) { if (tid < s) red[tid] += red[tid + s]; __syncthreads(); }
    float mean = red[0] * (1.0f / CT); __syncthreads();
    float d = (tid < CT) ? (r[tid] - mean) : 0.0f;
    red[tid] = d * d; __syncthreads();
    for (int s = 128; s > 0; s >>= 1) { if (tid < s) red[tid] += red[tid + s]; __syncthreads(); }
    float var = red[0] * (1.0f / CT); __syncthreads();
    if (tid < CT) outp[tid] = d * __frsqrt_rn(var + 1e-5f) * g[tid] + bb[tid];
    __syncthreads();
}

// ---------------------------------------------------------------------------
// K1: Mobile->Former attention + out proj + LN  ->  tok1 (ws)
// One block per batch element. 2 heads, dh=64, softmax over 1024 positions.
// ---------------------------------------------------------------------------
__global__ __launch_bounds__(256)
void bi_attn_kernel(const float* __restrict__ tokens, const float* __restrict__ feat,
                    const float* __restrict__ wq, const float* __restrict__ bq,
                    const float* __restrict__ wo, const float* __restrict__ bo,
                    const float* __restrict__ ln_g, const float* __restrict__ ln_b,
                    float* __restrict__ tok1) {
    const int b = blockIdx.x, tid = threadIdx.x;
    __shared__ float q_s[TN * CIN];
    __shared__ float ctx_s[TN * CIN];
    __shared__ float prob_s[HW];
    __shared__ float red_s[256];
    __shared__ float part_s[4 * 64];
    __shared__ float r_s[CT];

    // q = tokens @ wq^T + bq   (6 x 128)
    for (int idx = tid; idx < TN * CIN; idx += 256) {
        int t = idx >> 7, o = idx & 127;
        const float* tk = tokens + (t * BN + b) * CT;
        const float* w  = wq + o * CT;
        float acc = bq[o];
        for (int j = 0; j < CT; ++j) acc += tk[j] * w[j];
        q_s[idx] = acc;
    }
    __syncthreads();

    for (int h = 0; h < 2; ++h) {
        for (int t = 0; t < TN; ++t) {
            const float* qp = q_s + t * CIN + h * 64;
            float sc[4], lmax = -3.0e38f;
            for (int i = 0; i < 4; ++i) {
                int n = tid + i * 256;
                const float* kp = feat + ((size_t)(b * CIN + h * 64)) * HW + n;
                float acc = 0.0f;
                for (int d = 0; d < 64; ++d) acc += qp[d] * kp[(size_t)d * HW];
                sc[i] = acc * 0.125f;
                lmax = fmaxf(lmax, sc[i]);
            }
            red_s[tid] = lmax; __syncthreads();
            for (int s = 128; s > 0; s >>= 1) { if (tid < s) red_s[tid] = fmaxf(red_s[tid], red_s[tid + s]); __syncthreads(); }
            float bmax = red_s[0]; __syncthreads();
            float lsum = 0.0f;
            for (int i = 0; i < 4; ++i) {
                float e = __expf(sc[i] - bmax);
                prob_s[tid + i * 256] = e;
                lsum += e;
            }
            red_s[tid] = lsum; __syncthreads();
            for (int s = 128; s > 0; s >>= 1) { if (tid < s) red_s[tid] += red_s[tid + s]; __syncthreads(); }
            float inv = 1.0f / red_s[0]; __syncthreads();
            // ctx[d] = inv * sum_n prob[n] * feat[b][h*64+d][n]
            int d = tid & 63, part = tid >> 6;
            const float* kp = feat + ((size_t)(b * CIN + h * 64 + d)) * HW + part * 256;
            const float* pp = prob_s + part * 256;
            float acc = 0.0f;
            for (int n = 0; n < 256; ++n) acc += pp[n] * kp[n];
            part_s[part * 64 + d] = acc; __syncthreads();
            if (tid < 64)
                ctx_s[t * CIN + h * 64 + tid] =
                    inv * (part_s[tid] + part_s[64 + tid] + part_s[128 + tid] + part_s[192 + tid]);
            __syncthreads();
        }
    }

    // out proj + residual + LN per token
    for (int t = 0; t < TN; ++t) {
        if (tid < CT) {
            float acc = bo[tid];
            const float* cx = ctx_s + t * CIN;
            const float* w  = wo + tid * CIN;
            for (int j = 0; j < CIN; ++j) acc += cx[j] * w[j];
            r_s[tid] = tokens[(t * BN + b) * CT + tid] + acc;
        }
        __syncthreads();
        block_ln_192(r_s, red_s, tid, ln_g, ln_b, tok1 + (t * BN + b) * CT);
    }
}

// ---------------------------------------------------------------------------
// K2: token mixer self-attn + FFN + 2x LN -> tok (written to d_out tail)
// One block per batch. 4 heads, dm=48, 6 tokens.
// ---------------------------------------------------------------------------
__global__ __launch_bounds__(256)
void tm_kernel(const float* __restrict__ tok1,
               const float* __restrict__ wq, const float* __restrict__ bq,
               const float* __restrict__ wo, const float* __restrict__ bo,
               const float* __restrict__ ln1g, const float* __restrict__ ln1b,
               const float* __restrict__ w1, const float* __restrict__ b1,
               const float* __restrict__ w2, const float* __restrict__ b2,
               const float* __restrict__ ln2g, const float* __restrict__ ln2b,
               float* __restrict__ tok_out) {
    const int b = blockIdx.x, tid = threadIdx.x;
    __shared__ float tk[TN * CT];
    __shared__ float q_s[TN * CT];
    __shared__ float o_s[TN * CT];
    __shared__ float hm[TN * 384];
    __shared__ float red_s[256];

    for (int idx = tid; idx < TN * CT; idx += 256)
        tk[idx] = tok1[((idx / CT) * BN + b) * CT + (idx % CT)];
    __syncthreads();

    // q projection
    for (int idx = tid; idx < TN * CT; idx += 256) {
        int t = idx / CT, o = idx % CT;
        float acc = bq[o];
        const float* w = wq + o * CT;
        const float* x = tk + t * CT;
        for (int j = 0; j < CT; ++j) acc += x[j] * w[j];
        q_s[idx] = acc;
    }
    __syncthreads();

    // attention: thread per (t,h)
    if (tid < TN * 4) {
        int t = tid / 4, h = tid % 4;
        const float scale = 0.14433756729740643f; // 1/sqrt(48)
        float sc[TN], mx = -3.0e38f;
        for (int s = 0; s < TN; ++s) {
            float acc = 0.0f;
            for (int d = 0; d < 48; ++d) acc += q_s[t * CT + h * 48 + d] * tk[s * CT + h * 48 + d];
            sc[s] = acc * scale;
            mx = fmaxf(mx, sc[s]);
        }
        float sum = 0.0f;
        for (int s = 0; s < TN; ++s) { sc[s] = __expf(sc[s] - mx); sum += sc[s]; }
        float inv = 1.0f / sum;
        for (int d = 0; d < 48; ++d) {
            float acc = 0.0f;
            for (int s = 0; s < TN; ++s) acc += sc[s] * tk[s * CT + h * 48 + d];
            o_s[t * CT + h * 48 + d] = acc * inv;
        }
    }
    __syncthreads();

    // residual + out proj -> q_s (pre-LN)
    for (int idx = tid; idx < TN * CT; idx += 256) {
        int t = idx / CT, c = idx % CT;
        float acc = bo[c];
        const float* w = wo + c * CT;
        const float* x = o_s + t * CT;
        for (int j = 0; j < CT; ++j) acc += x[j] * w[j];
        q_s[idx] = tk[idx] + acc;
    }
    __syncthreads();
    for (int t = 0; t < TN; ++t)
        block_ln_192(q_s + t * CT, red_s, tid, ln1g, ln1b, tk + t * CT); // tokA -> tk

    // FFN
    for (int idx = tid; idx < TN * 384; idx += 256) {
        int t = idx / 384, u = idx % 384;
        float acc = b1[u];
        const float* w = w1 + u * CT;
        const float* x = tk + t * CT;
        for (int j = 0; j < CT; ++j) acc += x[j] * w[j];
        hm[idx] = 0.5f * acc * (1.0f + erff(acc * 0.70710678118654752f)); // exact gelu
    }
    __syncthreads();
    for (int idx = tid; idx < TN * CT; idx += 256) {
        int t = idx / CT, c = idx % CT;
        float acc = b2[c];
        const float* w = w2 + c * 384;
        const float* x = hm + t * 384;
        for (int u = 0; u < 384; ++u) acc += x[u] * w[u];
        q_s[idx] = tk[idx] + acc;
    }
    __syncthreads();
    for (int t = 0; t < TN; ++t)
        block_ln_192(q_s + t * CT, red_s, tid, ln2g, ln2b, tok_out + (t * BN + b) * CT);
}

// ---------------------------------------------------------------------------
// K3: gate MLPs from sel = tok[0]  ->  co1 (64x1024), co2 (64x1024), co3 (64x128)
// ---------------------------------------------------------------------------
__device__ __forceinline__
void gen_one(const float* sel_s, float* h_s, int tid,
             const float* w1, const float* b1, const float* w2, const float* b2,
             int outdim, float* co_row) {
    if (tid < 48) {
        float acc = b1[tid];
        const float* w = w1 + tid * CT;
        for (int j = 0; j < CT; ++j) acc += sel_s[j] * w[j];
        h_s[tid] = fmaxf(acc, 0.0f);
    }
    __syncthreads();
    for (int o = tid; o < outdim; o += 256) {
        float acc = b2[o];
        const float* w = w2 + o * 48;
        for (int u = 0; u < 48; ++u) acc += h_s[u] * w[u];
        co_row[o] = hsig(acc);
    }
    __syncthreads();
}

__global__ __launch_bounds__(256)
void gen_kernel(const float* __restrict__ tok,
                const float* g1w1, const float* g1b1, const float* g1w2, const float* g1b2,
                const float* g2w1, const float* g2b1, const float* g2w2, const float* g2b2,
                const float* g3w1, const float* g3b1, const float* g3w2, const float* g3b2,
                float* __restrict__ co1, float* __restrict__ co2, float* __restrict__ co3) {
    const int b = blockIdx.x, tid = threadIdx.x;
    __shared__ float sel[CT];
    __shared__ float h_s[48];
    if (tid < CT) sel[tid] = tok[b * CT + tid]; // tok[t=0][b][:]
    __syncthreads();
    gen_one(sel, h_s, tid, g1w1, g1b1, g1w2, g1b2, 1024, co1 + (size_t)b * 1024);
    gen_one(sel, h_s, tid, g2w1, g2b1, g2w2, g2b2, 1024, co2 + (size_t)b * 1024);
    gen_one(sel, h_s, tid, g3w1, g3b1, g3w2, g3b2, 128,  co3 + (size_t)b * 128);
}

// ---------------------------------------------------------------------------
// K4 / K6: WMMA bf16 pointwise-conv GEMM with fused BN + activation.
//   A  : [B][K][M=1024] bf16 (channel-major activations)
//   Wt : [N][K]         bf16 (conv weight, OI layout)
//   mode 0: BN + DyReLU(co: [B][2N]) -> bf16 out  (pw1)
//   mode 1: BN + 2*co   (co: [B][N]) -> f32  out  (pw2)
//
// Block = 256 threads (8 waves). Block tile: 128(M) x 128(N), K chunked by 128.
// Each wave owns 2 M-strips x 4 N-tiles (8 accumulators); each B fragment is
// reused by two WMMAs -> 12 ds_load_b128 per 8 v_wmma per K-step of 32.
//
// Fragment layouts per CDNA5 ISA 7.12.2 (wave32):
//   A 16x32 bf16 : lane -> M row (lane&15); element j -> K = j + 8*half + 8*(j>=8)
//                  => two contiguous 16B LDS reads per fragment.
//   B 32x16 bf16 : lane -> N col (lane&15); element j -> K = j + 8*half + 8*(j>=8)
//                  => two contiguous 16B LDS reads per fragment.
//   C/D f32      : element i -> M = i + 8*half, N = lane&15
//                  => 8 consecutive M per lane => vectorized 16B/32B stores.
// ---------------------------------------------------------------------------
union PackBf { uint4 q; bf16 e[8]; };
union PackF  { float4 f[2]; float e[8]; };

__global__ __launch_bounds__(256)
void wmma_pw_kernel(const bf16* __restrict__ A, const bf16* __restrict__ Wt,
                    const float* __restrict__ g, const float* __restrict__ bb,
                    const float* __restrict__ m_, const float* __restrict__ v_,
                    const float* __restrict__ co,
                    bf16* __restrict__ out_bf, float* __restrict__ out_f,
                    int K, int N, int mode) {
    const int M = HW;
    const int b  = blockIdx.z;
    const int m0 = blockIdx.x * 128;
    const int n0 = blockIdx.y * 128;
    const int tid  = threadIdx.x;
    const int lane = tid & 31;
    const int wv   = tid >> 5;
    const int half = (lane >> 4) & 1;
    const int l15  = lane & 15;
    const int LD   = 136; // row stride in bf16 (272B, 16B-aligned)

    __shared__ __align__(16) bf16 lds_a[128 * 136];
    __shared__ __align__(16) bf16 lds_b[128 * 136];

    v8f acc[2][4];
#pragma unroll
    for (int ms = 0; ms < 2; ++ms)
#pragma unroll
        for (int t = 0; t < 4; ++t)
            acc[ms][t] = (v8f){0.f,0.f,0.f,0.f,0.f,0.f,0.f,0.f};

    const int mgrp = wv & 3;            // 32-row group: strips 2*mgrp, 2*mgrp+1
    const int ngrp = (wv >> 2) * 64;    // 64-col group: 4 tiles of 16

    for (int kc = 0; kc < K; kc += 128) {
        __syncthreads();
        // stage A chunk: 128(M) x 128(K), coalesced over M, transpose into LDS
#pragma unroll 4
        for (int i = 0; i < 64; ++i) {
            int idx = i * 256 + tid;
            int mm = idx & 127, kk = idx >> 7;
            lds_a[mm * LD + kk] = A[((size_t)(b * K + kc + kk)) * M + m0 + mm];
        }
        // stage B chunk: 128(N) x 128(K), contiguous in K on both sides
#if USE_ASYNC_LDS
#pragma unroll
        for (int i = 0; i < 8; ++i) {
            int cidx = i * 256 + tid;              // 2048 chunks of 16B
            int nn = cidx >> 4, ck = (cidx & 15) * 8;
            __builtin_amdgcn_global_load_async_to_lds_b128(
                (gv4i_ptr)(bf16*)(Wt + (size_t)(n0 + nn) * K + kc + ck),
                (lv4i_ptr)(lds_b + nn * LD + ck), 0, 0);
        }
        __builtin_amdgcn_s_wait_asynccnt(0);
#else
#pragma unroll 4
        for (int i = 0; i < 64; ++i) {
            int idx = i * 256 + tid;
            int kk = idx & 127, nn = idx >> 7;
            lds_b[nn * LD + kk] = Wt[(size_t)(n0 + nn) * K + kc + kk];
        }
#endif
        __syncthreads();

#pragma unroll
        for (int kb = 0; kb < 4; ++kb) {
            v16bf af[2];
#pragma unroll
            for (int ms = 0; ms < 2; ++ms) {
                const bf16* ap = lds_a + (mgrp * 32 + ms * 16 + l15) * LD + kb * 32;
                uint4* aq = (uint4*)&af[ms];
                aq[0] = *(const uint4*)(ap + half * 8);        // j=0..7
                aq[1] = *(const uint4*)(ap + 16 + half * 8);   // j=8..15
            }
#pragma unroll
            for (int t = 0; t < 4; ++t) {
                v16bf bfv;
                const bf16* bp = lds_b + (ngrp + t * 16 + l15) * LD + kb * 32;
                uint4* bq = (uint4*)&bfv;
                bq[0] = *(const uint4*)(bp + half * 8);        // j=0..7
                bq[1] = *(const uint4*)(bp + 16 + half * 8);   // j=8..15
                acc[0][t] = __builtin_amdgcn_wmma_f32_16x16x32_bf16(
                    false, af[0], false, bfv, (short)0, acc[0][t], false, false);
                acc[1][t] = __builtin_amdgcn_wmma_f32_16x16x32_bf16(
                    false, af[1], false, bfv, (short)0, acc[1][t], false, false);
            }
        }
    }

    // fused epilogue: BN + DyReLU / scale, 8 consecutive M per lane -> b128
#pragma unroll
    for (int ms = 0; ms < 2; ++ms) {
        int mbase = m0 + mgrp * 32 + ms * 16 + half * 8;
#pragma unroll
        for (int t = 0; t < 4; ++t) {
            int ch = n0 + ngrp + t * 16 + l15;
            float scale = g[ch] * __frsqrt_rn(v_[ch] + 1e-5f);
            float mm = m_[ch], bbv = bb[ch];
            size_t base = ((size_t)(b * N + ch)) * M + mbase;
            if (mode == 0) {
                float ca = co[(size_t)b * (2 * N) + ch];
                float cb = co[(size_t)b * (2 * N) + N + ch];
                PackBf pk;
#pragma unroll
                for (int i = 0; i < 8; ++i) {
                    float y = (acc[ms][t][i] - mm) * scale + bbv;
                    float y1 = y * ((ca - 0.5f) * 2.0f + 1.0f);
                    float y2 = y * ((cb - 0.5f) * 2.0f);
                    pk.e[i] = (bf16)fmaxf(y1, y2);
                }
                *(uint4*)(out_bf + base) = pk.q;
            } else {
                float ca2 = co[(size_t)b * N + ch] * 2.0f;
                PackF pk;
#pragma unroll
                for (int i = 0; i < 8; ++i) {
                    float y = (acc[ms][t][i] - mm) * scale + bbv;
                    pk.e[i] = y * ca2;
                }
                *(float4*)(out_f + base)     = pk.f[0];
                *(float4*)(out_f + base + 4) = pk.f[1];
            }
        }
    }
}

// ---------------------------------------------------------------------------
// K5: depthwise 3x3 conv (pad 1) + BN + DyReLU. One block per (channel, batch).
// ---------------------------------------------------------------------------
__global__ __launch_bounds__(256)
void dw_kernel(const bf16* __restrict__ x1, const float* __restrict__ w,
               const float* __restrict__ g, const float* __restrict__ bb,
               const float* __restrict__ m_, const float* __restrict__ v_,
               const float* __restrict__ co2, bf16* __restrict__ x2) {
    const int c = blockIdx.x, b = blockIdx.y, tid = threadIdx.x;
    __shared__ float tile[34 * 34];
    const bf16* src = x1 + ((size_t)(b * 512 + c)) * HW;
    for (int idx = tid; idx < 34 * 34; idx += 256) {
        int iy = idx / 34 - 1, ix = idx % 34 - 1;
        float val = 0.0f;
        if ((unsigned)iy < 32u && (unsigned)ix < 32u) val = (float)src[iy * 32 + ix];
        tile[idx] = val;
    }
    __syncthreads();
    float wv[9];
#pragma unroll
    for (int i = 0; i < 9; ++i) wv[i] = w[c * 9 + i];
    float scale = g[c] * __frsqrt_rn(v_[c] + 1e-5f);
    float mm = m_[c], bbv = bb[c];
    float ca = co2[(size_t)b * 1024 + c];
    float cb = co2[(size_t)b * 1024 + 512 + c];
    bf16* dst = x2 + ((size_t)(b * 512 + c)) * HW;
    for (int p = tid; p < HW; p += 256) {
        int y = p >> 5, x = p & 31;
        float acc = 0.0f;
#pragma unroll
        for (int ky = 0; ky < 3; ++ky)
#pragma unroll
            for (int kx = 0; kx < 3; ++kx)
                acc += tile[(y + ky) * 34 + (x + kx)] * wv[ky * 3 + kx];
        float yv = (acc - mm) * scale + bbv;
        float y1 = yv * ((ca - 0.5f) * 2.0f + 1.0f);
        float y2 = yv * ((cb - 0.5f) * 2.0f);
        dst[p] = (bf16)fmaxf(y1, y2);
    }
}

// ---------------------------------------------------------------------------
// K7: Former->Mobile attention + final add: out = x3 + sp + feat (in-place on
// d_out's x3 region). One block per batch; 2 heads, softmax over 6 tokens.
// ---------------------------------------------------------------------------
__global__ __launch_bounds__(256)
void bo_kernel(const float* __restrict__ tok,
               const float* __restrict__ wk, const float* __restrict__ bk,
               const float* __restrict__ wv, const float* __restrict__ bv,
               const float* __restrict__ feat, float* __restrict__ xo) {
    const int b = blockIdx.x, tid = threadIdx.x;
    __shared__ float kk[CIN * TN]; // kk[c][t], c = h*64+d
    __shared__ float vv[CIN * TN]; // vv[c][t]
    for (int idx = tid; idx < CIN * TN; idx += 256) {
        int c = idx / TN, t = idx % TN;
        const float* x = tok + (t * BN + b) * CT;
        float ak = bk[c], av = bv[c];
        const float* rk = wk + c * CT;
        const float* rv = wv + c * CT;
        for (int j = 0; j < CT; ++j) { ak += x[j] * rk[j]; av += x[j] * rv[j]; }
        kk[c * TN + t] = ak;
        vv[c * TN + t] = av;
    }
    __syncthreads();

    const float* fb = feat + (size_t)b * CIN * HW;
    float* xb = xo + (size_t)b * CIN * HW;
    for (int n = tid; n < HW; n += 256) {
        float s[2][TN];
#pragma unroll
        for (int h = 0; h < 2; ++h)
            for (int t = 0; t < TN; ++t) s[h][t] = 0.0f;
        for (int c = 0; c < CIN; ++c) {
            float qv = xb[(size_t)c * HW + n];
            int h = c >> 6;
            const float* kr = kk + c * TN;
#pragma unroll
            for (int t = 0; t < TN; ++t) s[h][t] += qv * kr[t];
        }
        float a[2][TN];
#pragma unroll
        for (int h = 0; h < 2; ++h) {
            float mx = -3.0e38f;
#pragma unroll
            for (int t = 0; t < TN; ++t) { s[h][t] *= 0.125f; mx = fmaxf(mx, s[h][t]); }
            float sum = 0.0f;
#pragma unroll
            for (int t = 0; t < TN; ++t) { a[h][t] = __expf(s[h][t] - mx); sum += a[h][t]; }
            float inv = 1.0f / sum;
#pragma unroll
            for (int t = 0; t < TN; ++t) a[h][t] *= inv;
        }
        for (int c = 0; c < CIN; ++c) {
            int h = c >> 6;
            const float* vr = vv + c * TN;
            float sp = 0.0f;
#pragma unroll
            for (int t = 0; t < TN; ++t) sp += a[h][t] * vr[t];
            size_t o = (size_t)c * HW + n;
            xb[o] = xb[o] + sp + fb[o];
        }
    }
}

// ---------------------------------------------------------------------------
// Launch
// ---------------------------------------------------------------------------
extern "C" void kernel_launch(void* const* d_in, const int* in_sizes, int n_in,
                              void* d_out, int out_size, void* d_ws, size_t ws_size,
                              hipStream_t stream) {
    const float* feat   = (const float*)d_in[0];
    const float* tokens = (const float*)d_in[1];
    const float* bi_wq  = (const float*)d_in[2];
    const float* bi_bq  = (const float*)d_in[3];
    const float* bi_wo  = (const float*)d_in[4];
    const float* bi_bo  = (const float*)d_in[5];
    const float* bi_lng = (const float*)d_in[6];
    const float* bi_lnb = (const float*)d_in[7];
    const float* tm_wq  = (const float*)d_in[8];
    const float* tm_bq  = (const float*)d_in[9];
    const float* tm_wo  = (const float*)d_in[10];
    const float* tm_bo  = (const float*)d_in[11];
    const float* tm_l1g = (const float*)d_in[12];
    const float* tm_l1b = (const float*)d_in[13];
    const float* tm_w1  = (const float*)d_in[14];
    const float* tm_b1  = (const float*)d_in[15];
    const float* tm_w2  = (const float*)d_in[16];
    const float* tm_b2  = (const float*)d_in[17];
    const float* tm_l2g = (const float*)d_in[18];
    const float* tm_l2b = (const float*)d_in[19];
    const float* pw1_w  = (const float*)d_in[20];
    const float* bn1_g  = (const float*)d_in[21];
    const float* bn1_b  = (const float*)d_in[22];
    const float* bn1_m  = (const float*)d_in[23];
    const float* bn1_v  = (const float*)d_in[24];
    const float* g1_w1  = (const float*)d_in[25];
    const float* g1_b1  = (const float*)d_in[26];
    const float* g1_w2  = (const float*)d_in[27];
    const float* g1_b2  = (const float*)d_in[28];
    const float* dw_w   = (const float*)d_in[29];
    const float* bn2_g  = (const float*)d_in[30];
    const float* bn2_b  = (const float*)d_in[31];
    const float* bn2_m  = (const float*)d_in[32];
    const float* bn2_v  = (const float*)d_in[33];
    const float* g2_w1  = (const float*)d_in[34];
    const float* g2_b1  = (const float*)d_in[35];
    const float* g2_w2  = (const float*)d_in[36];
    const float* g2_b2  = (const float*)d_in[37];
    const float* pw2_w  = (const float*)d_in[38];
    const float* bn3_g  = (const float*)d_in[39];
    const float* bn3_b  = (const float*)d_in[40];
    const float* bn3_m  = (const float*)d_in[41];
    const float* bn3_v  = (const float*)d_in[42];
    const float* g3_w1  = (const float*)d_in[43];
    const float* g3_b1  = (const float*)d_in[44];
    const float* g3_w2  = (const float*)d_in[45];
    const float* g3_b2  = (const float*)d_in[46];
    const float* bo_wk  = (const float*)d_in[47];
    const float* bo_bk  = (const float*)d_in[48];
    const float* bo_wv  = (const float*)d_in[49];
    const float* bo_bv  = (const float*)d_in[50];

    float* out     = (float*)d_out;             // (64,128,32,32)
    float* tok_out = out + OUT_ELEMS;           // (6,64,192)

    // workspace layout (bytes, 256-aligned)
    char* ws = (char*)d_ws;
    bf16*  featbf = (bf16*)(ws + 0);            // 64*128*1024 bf16
    bf16*  wb1    = (bf16*)(ws + 16777216);     // 512*128 bf16
    bf16*  wb2    = (bf16*)(ws + 16908288);     // 128*512 bf16
    bf16*  x1     = (bf16*)(ws + 17039360);     // 64*512*1024 bf16
    bf16*  x2     = (bf16*)(ws + 84148224);     // 64*512*1024 bf16
    float* tok1   = (float*)(ws + 151257088);   // 6*64*192 f32
    float* co1    = (float*)(ws + 151552000);   // 64*1024 f32
    float* co2    = (float*)(ws + 151814144);   // 64*1024 f32
    float* co3    = (float*)(ws + 152076288);   // 64*128 f32

    // K0: bf16 conversions
    cvt_f32_bf16<<<4096, 256, 0, stream>>>(feat,  featbf, OUT_ELEMS);
    cvt_f32_bf16<<<64,   256, 0, stream>>>(pw1_w, wb1, 512 * 128);
    cvt_f32_bf16<<<64,   256, 0, stream>>>(pw2_w, wb2, 128 * 512);

    // K1: mobile -> former attention
    bi_attn_kernel<<<BN, 256, 0, stream>>>(tokens, feat, bi_wq, bi_bq, bi_wo, bi_bo,
                                           bi_lng, bi_lnb, tok1);
    // K2: token mixer + FFN (writes final tok)
    tm_kernel<<<BN, 256, 0, stream>>>(tok1, tm_wq, tm_bq, tm_wo, tm_bo, tm_l1g, tm_l1b,
                                      tm_w1, tm_b1, tm_w2, tm_b2, tm_l2g, tm_l2b, tok_out);
    // K3: gate MLPs
    gen_kernel<<<BN, 256, 0, stream>>>(tok_out, g1_w1, g1_b1, g1_w2, g1_b2,
                                       g2_w1, g2_b1, g2_w2, g2_b2,
                                       g3_w1, g3_b1, g3_w2, g3_b2, co1, co2, co3);
    // K4: pw1 GEMM (WMMA bf16) + BN1 + DyReLU -> x1 (bf16)
    wmma_pw_kernel<<<dim3(8, 4, BN), 256, 0, stream>>>(
        featbf, wb1, bn1_g, bn1_b, bn1_m, bn1_v, co1, x1, nullptr, 128, 512, 0);
    // K5: depthwise 3x3 + BN2 + DyReLU -> x2 (bf16)
    dw_kernel<<<dim3(512, BN), 256, 0, stream>>>(x1, dw_w, bn2_g, bn2_b, bn2_m, bn2_v,
                                                 co2, x2);
    // K6: pw2 GEMM (WMMA bf16) + BN3 + 2*gate -> x3 (f32, into d_out)
    wmma_pw_kernel<<<dim3(8, 1, BN), 256, 0, stream>>>(
        x2, wb2, bn3_g, bn3_b, bn3_m, bn3_v, co3, nullptr, out, 512, 128, 1);
    // K7: former -> mobile attention + out = x3 + sp + feat (in place)
    bo_kernel<<<BN, 256, 0, stream>>>(tok_out, bo_wk, bo_bk, bo_wv, bo_bv, feat, out);
}
